// DecoderBlock_59176059404572
// MI455X (gfx1250) — matrix operand order repaired and checked by reading, not compiled, not run on hardware
//
#include <hip/hip_runtime.h>
#include <hip/hip_bf16.h>
#include <math.h>

typedef __bf16 bf16;
typedef __attribute__((ext_vector_type(16))) __bf16 v16bf;
typedef __attribute__((ext_vector_type(8)))  __bf16 v8bf;
typedef __attribute__((ext_vector_type(4)))  __bf16 v4bf;
typedef __attribute__((ext_vector_type(8)))  float  v8f;

#define DEV static __device__ __forceinline__

#if defined(__has_builtin)
#if __has_builtin(__builtin_amdgcn_tensor_load_to_lds)
#define USE_TDM 1
#endif
#endif

DEV int lane_id() { return (int)(threadIdx.x & 31u); }

// Low 32 bits of a flat LDS address == logical LDS byte address (aperture rule)
DEV unsigned lds_off(const void* p) { return (unsigned)(unsigned long long)p; }

// ---------------------------------------------------------------------------
// CDNA5 async global->LDS copy (ASYNCcnt), GV addressing mode
DEV void async_copy_b128(unsigned lds, const void* gaddr) {
  asm volatile("global_load_async_to_lds_b128 %0, %1, off"
               :: "v"(lds), "v"((unsigned long long)gaddr) : "memory");
}
DEV void wait_async0() { asm volatile("s_wait_asynccnt 0x0" ::: "memory"); }
DEV void wait_async1() { asm volatile("s_wait_asynccnt 0x1" ::: "memory"); }

#ifdef USE_TDM
typedef __attribute__((ext_vector_type(4))) unsigned v4u;
typedef __attribute__((ext_vector_type(4))) int v4i;
typedef __attribute__((ext_vector_type(8))) int v8i;

// TDM 2D tile load: tile_rows x tile_elems (bf16) from row-major tensor
// (rows x row_elems, row stride = stride_elems) into LDS at byte addr `lds`.
// Descriptor per CDNA5 ISA 8.3/8.4 (count=1, type=2 "image", data_size=2B).
DEV void tdm_load_2d(unsigned lds, const void* gaddr, unsigned rows,
                     unsigned row_elems, unsigned tile_rows, unsigned tile_elems,
                     unsigned stride_elems) {
  const unsigned long long ga = (unsigned long long)gaddr;
  v4u g0;
  g0[0] = 1u;                                                // count=1 (user D#)
  g0[1] = lds;                                               // lds_addr (bytes)
  g0[2] = (unsigned)ga;                                      // global_addr[31:0]
  g0[3] = (unsigned)((ga >> 32) & 0x01FFFFFFu) | (2u << 30); // ga[56:32] | type=2
  v8i g1;
  g1[0] = (int)(1u << 16);                                   // data_size=1 (2 B)
  g1[1] = (int)((row_elems & 0xFFFFu) << 16);                // tensor_dim0 lo
  g1[2] = (int)((row_elems >> 16) | ((rows & 0xFFFFu) << 16));       // d0 hi|d1 lo
  g1[3] = (int)((rows >> 16) | ((tile_elems & 0xFFFFu) << 16));      // d1 hi|tile0
  g1[4] = (int)(tile_rows & 0xFFFFu);                        // tile_dim1 (dim2=0)
  g1[5] = (int)stride_elems;                                 // dim0_stride lo
  g1[6] = 0;
  g1[7] = 0;
  v4i gz = {0, 0, 0, 0};
#if __clang_major__ >= 23
  v8i gz8 = {0, 0, 0, 0, 0, 0, 0, 0};
  __builtin_amdgcn_tensor_load_to_lds(g0, g1, gz, gz, gz8, 0);
#else
  __builtin_amdgcn_tensor_load_to_lds(g0, g1, gz, gz, 0);
#endif
}
#endif

// ---------------------------------------------------------------------------
// WMMA fragment loaders (CDNA5 ISA 7.12.2 layouts, wave32)
// A-matrix 16x32 bf16: lanes 0-15 row M=lane, K {0..7, 16..23};
//                      lanes 16-31 row M=lane-16, K {8..15, 24..31}
DEV v16bf load_a16x32(const bf16* tile, int ld) {
  const int l = lane_id();
  const int m = l & 15;
  const int h = l >> 4;
  const bf16* p = tile + (size_t)m * ld + h * 8;
  v8bf x0 = *(const v8bf*)(p);
  v8bf x1 = *(const v8bf*)(p + 16);
  v16bf r;
#pragma unroll
  for (int i = 0; i < 8; ++i) { r[i] = x0[i]; r[i + 8] = x1[i]; }
  return r;
}

// B-matrix 32x16 bf16 from row-major Wt[N x K] (B_mat[k][n] = Wt[n][k]):
// lanes 0-15 col N=lane, K {0..7, 16..23}; lanes 16-31 col N=lane-16, K {8..15, 24..31}
DEV v16bf load_b32x16(const bf16* wtile, int ld) {
  const int l = lane_id();
  const int n = l & 15;
  const int h = l >> 4;
  const bf16* p = wtile + (size_t)n * ld + h * 8;
  v8bf x0 = *(const v8bf*)(p);
  v8bf x1 = *(const v8bf*)(p + 16);
  v16bf r;
#pragma unroll
  for (int i = 0; i < 8; ++i) { r[i] = x0[i]; r[i + 8] = x1[i]; }
  return r;
}

DEV v8f wmma_bf16(v16bf a, v16bf b, v8f c) {
  return __builtin_amdgcn_wmma_f32_16x16x32_bf16(false, a, false, b,
                                                 (short)0, c, false, false);
}

DEV float gelu_tanh(float x) {
  const float x3 = x * x * x;
  return 0.5f * x * (1.0f + tanhf(0.7978845608028654f * (x + 0.044715f * x3)));
}

// ---------------------------------------------------------------------------
// fp32 -> bf16 elementwise (weights)
__global__ __launch_bounds__(256) void f2bf_kernel(const float* __restrict__ s,
                                                   bf16* __restrict__ d, int n) {
  const int i = (int)(blockIdx.x * blockDim.x + threadIdx.x) * 4;
  if (i + 3 < n) {
    const float4 v = *(const float4*)(s + i);
    v4bf o;
    o[0] = (bf16)v.x; o[1] = (bf16)v.y; o[2] = (bf16)v.z; o[3] = (bf16)v.w;
    *(v4bf*)(d + i) = o;
  }
}

// ---------------------------------------------------------------------------
// LayerNorm over D=1024, one row per 256-thread block; writes fp32 + bf16
__global__ __launch_bounds__(256) void ln_kernel(const float* __restrict__ x,
                                                 const float* __restrict__ g,
                                                 const float* __restrict__ b,
                                                 float* __restrict__ yf,
                                                 bf16* __restrict__ yh) {
  const int row = blockIdx.x;
  const int t = threadIdx.x;
  const float4 v = *(const float4*)(x + (size_t)row * 1024 + t * 4);
  float s = v.x + v.y + v.z + v.w;
  float q = v.x * v.x + v.y * v.y + v.z * v.z + v.w * v.w;
#pragma unroll
  for (int off = 1; off < 32; off <<= 1) {
    s += __shfl_xor(s, off);
    q += __shfl_xor(q, off);
  }
  __shared__ float ss[8], sq[8];
  const int w = t >> 5;
  if ((t & 31) == 0) { ss[w] = s; sq[w] = q; }
  __syncthreads();
  float S = 0.f, Q = 0.f;
#pragma unroll
  for (int i = 0; i < 8; ++i) { S += ss[i]; Q += sq[i]; }
  const float mean = S * (1.0f / 1024.0f);
  const float var  = Q * (1.0f / 1024.0f) - mean * mean;
  const float inv  = rsqrtf(var + 1e-5f);
  const float4 gv = *(const float4*)(g + t * 4);
  const float4 bv = *(const float4*)(b + t * 4);
  float4 o;
  o.x = (v.x - mean) * inv * gv.x + bv.x;
  o.y = (v.y - mean) * inv * gv.y + bv.y;
  o.z = (v.z - mean) * inv * gv.z + bv.z;
  o.w = (v.w - mean) * inv * gv.w + bv.w;
  *(float4*)(yf + (size_t)row * 1024 + t * 4) = o;
  v4bf oh;
  oh[0] = (bf16)o.x; oh[1] = (bf16)o.y; oh[2] = (bf16)o.z; oh[3] = (bf16)o.w;
  *(v4bf*)(yh + (size_t)row * 1024 + t * 4) = oh;
}

// ---------------------------------------------------------------------------
// Fused per-head QKV projection. Per-head 64x64 weight tiles staged into LDS
// by the Tensor Data Mover (wave 0 issues 3 descriptors, TENSORcnt wait),
// then all 8 waves consume them as WMMA B-fragments via ds loads.
// Fragment loads are grouped ahead of the WMMAs so multiple ds loads stay
// in flight (partial s_wait_dscnt) instead of load->wait0->wmma chains.
__global__ __launch_bounds__(256) void qkv_kernel(const bf16* __restrict__ x1h,
                                                  const bf16* __restrict__ Wqh,
                                                  const bf16* __restrict__ Wkh,
                                                  const bf16* __restrict__ Wvh,
                                                  bf16* __restrict__ Qo,
                                                  bf16* __restrict__ Ko,
                                                  bf16* __restrict__ Vt) {
  __shared__ __align__(16) bf16 wq_s[64 * 64];
  __shared__ __align__(16) bf16 wk_s[64 * 64];
  __shared__ __align__(16) bf16 wv_s[64 * 64];
  const int tid = (int)threadIdx.x;
  const int w  = tid >> 5;
  const int h  = blockIdx.y;
  const int m0 = blockIdx.x * 128 + w * 16;   // row in flattened [B*S)
  const int b  = m0 >> 11;
  const int s0 = m0 & 2047;
  const bf16* Ab = x1h + (size_t)m0 * 1024 + h * 64;
  const bf16* Wq = Wqh + (size_t)h * 4096;
  const bf16* Wk = Wkh + (size_t)h * 4096;
  const bf16* Wv = Wvh + (size_t)h * 4096;

#ifdef USE_TDM
  if (w == 0) {                // one wave issues the TDM descriptors
    tdm_load_2d(lds_off(wq_s), Wq, 64, 64, 64, 64, 64);
    tdm_load_2d(lds_off(wk_s), Wk, 64, 64, 64, 64, 64);
    tdm_load_2d(lds_off(wv_s), Wv, 64, 64, 64, 64, 64);
    __builtin_amdgcn_s_wait_tensorcnt(0);
  }
#else
#pragma unroll
  for (int i = tid; i < 512; i += 256) {      // cooperative fallback copy
    ((v8bf*)wq_s)[i] = ((const v8bf*)Wq)[i];
    ((v8bf*)wk_s)[i] = ((const v8bf*)Wk)[i];
    ((v8bf*)wv_s)[i] = ((const v8bf*)Wv)[i];
  }
#endif
  __syncthreads();

  v8f aq[4] = {}, ak[4] = {}, av[4] = {};
#pragma unroll
  for (int kc = 0; kc < 64; kc += 32) {
    const v16bf a = load_a16x32(Ab + kc, 1024);
    v16bf bq[4], bk[4], bv4[4];
#pragma unroll
    for (int t4 = 0; t4 < 4; ++t4)
      bq[t4] = load_b32x16(wq_s + (size_t)(t4 * 16) * 64 + kc, 64);
#pragma unroll
    for (int t4 = 0; t4 < 4; ++t4) aq[t4] = wmma_bf16(a, bq[t4], aq[t4]);
#pragma unroll
    for (int t4 = 0; t4 < 4; ++t4)
      bk[t4] = load_b32x16(wk_s + (size_t)(t4 * 16) * 64 + kc, 64);
#pragma unroll
    for (int t4 = 0; t4 < 4; ++t4) ak[t4] = wmma_bf16(a, bk[t4], ak[t4]);
#pragma unroll
    for (int t4 = 0; t4 < 4; ++t4)
      bv4[t4] = load_b32x16(wv_s + (size_t)(t4 * 16) * 64 + kc, 64);
#pragma unroll
    for (int t4 = 0; t4 < 4; ++t4) av[t4] = wmma_bf16(a, bv4[t4], av[t4]);
  }
  const int l = lane_id(), n = l & 15, half = l >> 4;
  const size_t bh = (size_t)(b * 16 + h);
  bf16* Qb = Qo + (bh * 2048 + s0) * 64;
  bf16* Kb = Ko + (bh * 2048 + s0) * 64;
  bf16* Vb = Vt + bh * 64 * 2048;
#pragma unroll
  for (int t4 = 0; t4 < 4; ++t4) {
    const int e = t4 * 16 + n;
    v8bf vv;
#pragma unroll
    for (int r = 0; r < 8; ++r) {
      const int sl = half * 8 + r;
      Qb[(size_t)sl * 64 + e] = (bf16)(aq[t4][r] * 0.125f);  // 1/sqrt(DH)
      Kb[(size_t)sl * 64 + e] = (bf16)(ak[t4][r]);
      vv[r] = (bf16)(av[t4][r]);
    }
    *(v8bf*)(Vb + (size_t)e * 2048 + s0 + half * 8) = vv;    // transposed store
  }
}

// ---------------------------------------------------------------------------
// Causal flash attention: one 16-row q-tile per wave, 32-wide k-chunks,
// fp32 online softmax, P transposed through per-wave LDS tile for P@V WMMA.
// K-fragments hoisted ahead of score WMMAs; V-fragment global loads issued
// before the softmax VALU work so their latency hides under exp/reductions.
__global__ __launch_bounds__(256) void flash_kernel(const bf16* __restrict__ Q,
                                                    const bf16* __restrict__ K,
                                                    const bf16* __restrict__ Vt,
                                                    bf16* __restrict__ attn) {
  __shared__ __align__(16) bf16 pl[8][16][32];
  const int w  = (int)(threadIdx.x >> 5);
  const int t  = blockIdx.x * 8 + w;           // tile over (b,h,qtile)
  const int qt = t & 127;
  const int h  = (t >> 7) & 15;
  const int b  = t >> 11;
  const int s0 = qt * 16;
  const size_t bh = (size_t)(b * 16 + h);
  const bf16* Qb = Q + (bh * 2048 + s0) * 64;
  const bf16* Kb = K + bh * 2048 * 64;
  const bf16* Vb = Vt + bh * 64 * 2048;
  const int l = lane_id(), n = l & 15, half = l >> 4;

  const v16bf aq0 = load_a16x32(Qb, 64);       // d = 0..31
  const v16bf aq1 = load_a16x32(Qb + 32, 64);  // d = 32..63

  float mr[8], lr[8];
  v8f o[4] = {};
#pragma unroll
  for (int r = 0; r < 8; ++r) { mr[r] = -1e30f; lr[r] = 0.f; }

  const int nch = (s0 + 47) >> 5;              // k-chunks covering k <= s0+15
  for (int c = 0; c < nch; ++c) {
    const int kb = c * 32;
    v16bf bk0 = load_b32x16(Kb + (size_t)kb * 64, 64);
    v16bf bk1 = load_b32x16(Kb + (size_t)kb * 64 + 32, 64);
    v16bf bk2 = load_b32x16(Kb + (size_t)(kb + 16) * 64, 64);
    v16bf bk3 = load_b32x16(Kb + (size_t)(kb + 16) * 64 + 32, 64);
    v8f sa = {}, sb = {};
    sa = wmma_bf16(aq0, bk0, sa);
    sa = wmma_bf16(aq1, bk1, sa);
    sb = wmma_bf16(aq0, bk2, sb);
    sb = wmma_bf16(aq1, bk3, sb);

    v16bf bv[4];                               // V frags: hide under softmax
#pragma unroll
    for (int t4 = 0; t4 < 4; ++t4)
      bv[t4] = load_b32x16(Vb + (size_t)(t4 * 16) * 2048 + kb, 2048);

    if (kb + 31 > s0) {                        // diagonal chunk: causal mask
#pragma unroll
      for (int r = 0; r < 8; ++r) {
        const int qg = s0 + half * 8 + r;
        if (kb + n > qg)      sa[r] = -1e30f;
        if (kb + 16 + n > qg) sb[r] = -1e30f;
      }
    }

    float rm[8], rs[8];
#pragma unroll
    for (int r = 0; r < 8; ++r) rm[r] = fmaxf(sa[r], sb[r]);
#pragma unroll
    for (int off = 1; off < 16; off <<= 1)
#pragma unroll
      for (int r = 0; r < 8; ++r) rm[r] = fmaxf(rm[r], __shfl_xor(rm[r], off));

#pragma unroll
    for (int r = 0; r < 8; ++r) {
      const float mn = fmaxf(mr[r], rm[r]);
      const float cf = __expf(mr[r] - mn);
      mr[r] = mn;
      const float p0 = __expf(sa[r] - mn);
      const float p1 = __expf(sb[r] - mn);
      rs[r] = p0 + p1;
      lr[r] *= cf;
      const int ml = half * 8 + r;
      pl[w][ml][n]      = (bf16)p0;            // C-layout -> row-major LDS
      pl[w][ml][16 + n] = (bf16)p1;
#pragma unroll
      for (int t4 = 0; t4 < 4; ++t4) o[t4][r] *= cf;
    }
#pragma unroll
    for (int off = 1; off < 16; off <<= 1)
#pragma unroll
      for (int r = 0; r < 8; ++r) rs[r] += __shfl_xor(rs[r], off);
#pragma unroll
    for (int r = 0; r < 8; ++r) lr[r] += rs[r];

    const v16bf ap = load_a16x32(&pl[w][0][0], 32);  // A-layout reload (ds)
#pragma unroll
    for (int t4 = 0; t4 < 4; ++t4) o[t4] = wmma_bf16(ap, bv[t4], o[t4]);
  }

  bf16* outp = attn + (size_t)(b * 2048 + s0) * 1024 + h * 64;
#pragma unroll
  for (int t4 = 0; t4 < 4; ++t4) {
    const int e = t4 * 16 + n;
#pragma unroll
    for (int r = 0; r < 8; ++r) {
      const int sl = half * 8 + r;
      outp[(size_t)sl * 1024 + e] = (bf16)(o[t4][r] / lr[r]);
    }
  }
}

// ---------------------------------------------------------------------------
// C[M,N] = A[M,K] @ W[N,K]^T (+bias)(+gelu)(+addf residual) -> fp32 and/or bf16
// Block: 256 threads = 8 waves; wave computes 32x64 via 8 f32 accumulators.
// Shared 64-row W chunk double-buffered in LDS via async global->LDS copies
// (one GLOBAL_LOAD_ASYNC_TO_LDS_B128 per thread per chunk), synchronized with
// s_wait_asynccnt + workgroup barrier. A-fragment global loads are issued
// before the async wait so they overlap the DMA; all 4 B-fragments are loaded
// before the 8 WMMAs so ds loads drain with partial waits.
__global__ __launch_bounds__(256) void gemm_kernel(const bf16* __restrict__ A,
                                                   const bf16* __restrict__ W,
                                                   const float* __restrict__ bias,
                                                   const float* __restrict__ addf,
                                                   float* __restrict__ outf,
                                                   bf16* __restrict__ outh,
                                                   int N, int K, int act) {
  __shared__ __align__(16) bf16 wbuf[2][64][32];
  const int tid = (int)threadIdx.x;
  const int w   = tid >> 5;
  const int m0  = blockIdx.x * 256 + w * 32;
  const int n0  = blockIdx.y * 64;
  const bf16* A0 = A + (size_t)m0 * K;
  const bf16* A1 = A + (size_t)(m0 + 16) * K;
  const bf16* W0 = W + (size_t)n0 * K;
  const int crow = tid >> 2;            // 64 rows, 4 threads per row
  const int cseg = (tid & 3) * 8;       // 8 bf16 (16 B) per thread
  v8f acc[2][4] = {};

  async_copy_b128(lds_off(&wbuf[0][crow][cseg]),
                  W0 + (size_t)crow * K + cseg);          // stage chunk 0
  const int nc = K / 32;
  for (int c = 0; c < nc; ++c) {
    const int k0 = c * 32;
    const v16bf a0 = load_a16x32(A0 + k0, K);   // overlap with async DMA wait
    const v16bf a1 = load_a16x32(A1 + k0, K);
    if (c + 1 < nc) {
      async_copy_b128(lds_off(&wbuf[(c + 1) & 1][crow][cseg]),
                      W0 + (size_t)crow * K + k0 + 32 + cseg);
      wait_async1();                    // oldest copy (chunk c) complete
    } else {
      wait_async0();
    }
    __syncthreads();                    // chunk c visible to all waves
    const bf16* wb = &wbuf[c & 1][0][0];
    v16bf bfr[4];
#pragma unroll
    for (int t4 = 0; t4 < 4; ++t4)
      bfr[t4] = load_b32x16(wb + (size_t)(t4 * 16) * 32, 32);
#pragma unroll
    for (int t4 = 0; t4 < 4; ++t4) {
      acc[0][t4] = wmma_bf16(a0, bfr[t4], acc[0][t4]);
      acc[1][t4] = wmma_bf16(a1, bfr[t4], acc[1][t4]);
    }
    __syncthreads();                    // done reading buf before it is refilled
  }

  const int l = lane_id(), n = l & 15, half = l >> 4;
#pragma unroll
  for (int i = 0; i < 2; ++i)
#pragma unroll
    for (int t4 = 0; t4 < 4; ++t4) {
      const int col = n0 + t4 * 16 + n;
      const float bv = bias ? bias[col] : 0.0f;
#pragma unroll
      for (int r = 0; r < 8; ++r) {
        const int row = m0 + i * 16 + half * 8 + r;
        float v = acc[i][t4][r] + bv;
        if (act) v = gelu_tanh(v);
        if (addf) v += addf[(size_t)row * N + col];
        if (outf) outf[(size_t)row * N + col] = v;
        if (outh) outh[(size_t)row * N + col] = (bf16)v;
      }
    }
}

// ---------------------------------------------------------------------------
extern "C" void kernel_launch(void* const* d_in, const int* in_sizes, int n_in,
                              void* d_out, int out_size, void* d_ws, size_t ws_size,
                              hipStream_t stream) {
  (void)in_sizes; (void)n_in; (void)out_size; (void)ws_size;
  const float* x    = (const float*)d_in[0];
  const float* ln1g = (const float*)d_in[1];
  const float* ln1b = (const float*)d_in[2];
  const float* Wq   = (const float*)d_in[3];
  const float* Wk   = (const float*)d_in[4];
  const float* Wv   = (const float*)d_in[5];
  const float* Wo   = (const float*)d_in[6];
  const float* ln2g = (const float*)d_in[7];
  const float* ln2b = (const float*)d_in[8];
  const float* W1   = (const float*)d_in[9];
  const float* b1   = (const float*)d_in[10];
  const float* W2   = (const float*)d_in[11];
  const float* b2   = (const float*)d_in[12];

  constexpr int D = 1024, H = 16, DH = 64;
  constexpr size_t R = 4096;  // B*S rows

  char* ws = (char*)d_ws;
  size_t off = 0;
  auto alloc = [&](size_t bytes) -> void* {
    void* p = ws + off;
    off += (bytes + 255) & ~(size_t)255;
    return p;
  };
  float* x1f = (float*)alloc(R * D * 4);
  bf16*  x1h = (bf16*)alloc(R * D * 2);
  bf16*  qh  = (bf16*)alloc(R * D * 2);
  bf16*  kh  = (bf16*)alloc(R * D * 2);
  bf16*  vth = (bf16*)alloc(R * D * 2);       // V transposed [B,H,DH,S]
  bf16*  ah  = (bf16*)alloc(R * D * 2);       // attention output
  float* u2f = (float*)alloc(R * D * 4);
  bf16*  u2h = (bf16*)alloc(R * D * 2);
  bf16*  ffh = (bf16*)alloc(R * 4 * D * 2);   // FFN hidden
  bf16*  Wqh = (bf16*)alloc((size_t)H * DH * DH * 2);
  bf16*  Wkh = (bf16*)alloc((size_t)H * DH * DH * 2);
  bf16*  Wvh = (bf16*)alloc((size_t)H * DH * DH * 2);
  bf16*  Woh = (bf16*)alloc((size_t)D * D * 2);
  bf16*  W1h = (bf16*)alloc((size_t)4 * D * D * 2);
  bf16*  W2h = (bf16*)alloc((size_t)4 * D * D * 2);
  float* uf  = (float*)d_out;                 // reuse d_out as u scratch

  auto cvt = [&](const float* s, bf16* d, int nelem) {
    f2bf_kernel<<<dim3((unsigned)((nelem / 4 + 255) / 256)), dim3(256), 0, stream>>>(s, d, nelem);
  };
  cvt(Wq, Wqh, H * DH * DH);
  cvt(Wk, Wkh, H * DH * DH);
  cvt(Wv, Wvh, H * DH * DH);
  cvt(Wo, Woh, D * D);
  cvt(W1, W1h, 4 * D * D);
  cvt(W2, W2h, 4 * D * D);

  // x1 = LN1(x)
  ln_kernel<<<dim3((unsigned)R), dim3(256), 0, stream>>>(x, ln1g, ln1b, x1f, x1h);
  // per-head q,k,v projections (TDM-staged weights)
  qkv_kernel<<<dim3((unsigned)(R / 128), H), dim3(256), 0, stream>>>(x1h, Wqh, Wkh, Wvh, qh, kh, vth);
  // causal flash attention: B*H*(S/16)/8 = 512 blocks
  flash_kernel<<<dim3(512), dim3(256), 0, stream>>>(qh, kh, vth, ah);
  // u = x1 + attn @ Wo^T
  gemm_kernel<<<dim3((unsigned)(R / 256), D / 64), dim3(256), 0, stream>>>(
      ah, Woh, nullptr, x1f, uf, nullptr, D, D, 0);
  // u2 = LN2(u)
  ln_kernel<<<dim3((unsigned)R), dim3(256), 0, stream>>>(uf, ln2g, ln2b, u2f, u2h);
  // h = gelu(u2 @ W1^T + b1)
  gemm_kernel<<<dim3((unsigned)(R / 256), 4 * D / 64), dim3(256), 0, stream>>>(
      u2h, W1h, b1, nullptr, nullptr, ffh, 4 * D, D, 1);
  // out = h @ W2^T + b2 + u2
  gemm_kernel<<<dim3((unsigned)(R / 256), D / 64), dim3(256), 0, stream>>>(
      ffh, W2h, b2, u2f, (float*)d_out, nullptr, D, 4 * D, 0);
}
